// NNFMLoss_49727131353722
// MI455X (gfx1250) — matrix operand order repaired and checked by reading, not compiled, not run on hardware
//
#include <hip/hip_runtime.h>
#include <math.h>

#define HW 16384
#define CCH 768
#define BI 64      // i-rows per block
#define BJ 256     // j-cols per block tile
#define BK 64      // k-chunk
#define BKP 80     // padded LDS row stride (halves): 160B, 16B aligned
#define EPSF 1e-8f

typedef __attribute__((ext_vector_type(16))) __bf16 v16bf;
typedef __attribute__((ext_vector_type(8)))  float  v8f;
typedef __attribute__((ext_vector_type(4)))  unsigned int v4u;

union FragU { v4u u[2]; v16bf v; };

// round-to-nearest-even float -> bf16 bit pattern
static __device__ __forceinline__ unsigned short f2bf(float f) {
    unsigned int u = __float_as_uint(f);
    u = (u + 0x7FFFu + ((u >> 16) & 1u)) >> 16;
    return (unsigned short)u;
}

// ---------------- kernel 1: column norms of style features ----------------
__global__ __launch_bounds__(256) void bnorm_kernel(const float* __restrict__ b,
                                                    float* __restrict__ bn) {
    int j = blockIdx.x * 256 + threadIdx.x;
    float s = 0.f;
    for (int c = 0; c < CCH; ++c) {
        float v = b[(size_t)c * HW + j];
        s += v * v;
    }
    bn[j] = sqrtf(s + EPSF);
}

// ------------- kernel 2: transpose [C][HW] f32 -> [HW][C] bf16 -------------
__global__ __launch_bounds__(256) void transpose_bf16_kernel(
        const float* __restrict__ src, const float* __restrict__ norm,
        int use_norm, unsigned short* __restrict__ dst) {
    __shared__ float tile[32][33];
    const int n0 = blockIdx.x * 32;
    const int c0 = blockIdx.y * 32;
    const int tn = threadIdx.x & 31;
    const int tc = threadIdx.x >> 5;  // 0..7
#pragma unroll
    for (int p = 0; p < 4; ++p) {
        int cc = tc + p * 8;
        tile[cc][tn] = src[(size_t)(c0 + cc) * HW + n0 + tn];
    }
    __syncthreads();
#pragma unroll
    for (int p = 0; p < 4; ++p) {
        int nn = tc + p * 8;
        float sc = use_norm ? (1.f / (norm[n0 + nn] + EPSF)) : 1.f;
        dst[(size_t)(n0 + nn) * CCH + c0 + tn] = f2bf(tile[tn][nn] * sc);
    }
}

// ------- kernel 3: fused (A^T * B_n) GEMM + row-argmax, bf16 WMMA ---------
// block tile: 64(i) x 256(j); 8 waves as 2(i) x 4(j); wave tile 32x64 = 2x4 WMMA tiles
__global__ __launch_bounds__(256) void gemm_argmax_kernel(
        const unsigned short* __restrict__ At,   // [HW][C] bf16 (x, raw)
        const unsigned short* __restrict__ Bt,   // [HW][C] bf16 (s, normalized)
        int* __restrict__ zbest) {
    __shared__ unsigned short As[BI][BKP];       // 10 KB
    __shared__ unsigned short Bs[BJ][BKP];       // 40 KB
    __shared__ float mV[4][BI];
    __shared__ int   mI[4][BI];

    const int tid = threadIdx.x;
    const int lane = tid & 31;
    const int wid = tid >> 5;
    const int waveRow = wid >> 2;   // 0..1 : i offset 32*waveRow
    const int waveCol = wid & 3;    // 0..3 : j offset 64*waveCol
    const int N = lane & 15;
    const int half = lane >> 4;
    const int i0 = blockIdx.x * BI;

    float bestv[16];
    int   besti[16];
#pragma unroll
    for (int t = 0; t < 16; ++t) { bestv[t] = -3.0e38f; besti[t] = 0; }

    const v8f z8 = {0.f, 0.f, 0.f, 0.f, 0.f, 0.f, 0.f, 0.f};

    for (int jt = 0; jt < HW; jt += BJ) {
        v8f acc[2][4];
#pragma unroll
        for (int ti = 0; ti < 2; ++ti)
#pragma unroll
            for (int tj = 0; tj < 4; ++tj) acc[ti][tj] = z8;

        for (int kt = 0; kt < CCH; kt += BK) {
            __syncthreads();
            // stage A: 64 rows x 64 halves = 512 x 16B chunks (2 per thread)
#pragma unroll
            for (int q = 0; q < 2; ++q) {
                int idx = q * 256 + tid;
                int r = idx >> 3, c8 = (idx & 7) << 3;
                *(v4u*)&As[r][c8] =
                    *(const v4u*)&At[(size_t)(i0 + r) * CCH + kt + c8];
            }
            // stage B: 256 rows x 64 halves = 2048 x 16B chunks (8 per thread)
#pragma unroll
            for (int q = 0; q < 8; ++q) {
                int idx = q * 256 + tid;
                int r = idx >> 3, c8 = (idx & 7) << 3;
                *(v4u*)&Bs[r][c8] =
                    *(const v4u*)&Bt[(size_t)(jt + r) * CCH + kt + c8];
            }
            __syncthreads();

#pragma unroll
            for (int s = 0; s < 2; ++s) {   // two K=32 substeps
                const int kb = s * 32;
                FragU fa[2], fb[4];
#pragma unroll
                for (int ti = 0; ti < 2; ++ti) {
                    // A 16x32 layout: e<8 -> K=kb+8*half+e ; e>=8 -> K=kb+16+8*half+(e-8)
                    int arow = waveRow * 32 + ti * 16 + N;
                    int ac = kb + (half << 3);
                    fa[ti].u[0] = *(const v4u*)&As[arow][ac];
                    fa[ti].u[1] = *(const v4u*)&As[arow][ac + 16];
                }
#pragma unroll
                for (int tj = 0; tj < 4; ++tj) {
                    // B 32x16 layout: e<8 -> K=kb+8*half+e ; e>=8 -> K=kb+16+8*half+(e-8)
                    int brow = waveCol * 64 + tj * 16 + N;
                    int bc = kb + (half << 3);
                    fb[tj].u[0] = *(const v4u*)&Bs[brow][bc];
                    fb[tj].u[1] = *(const v4u*)&Bs[brow][bc + 16];
                }
#pragma unroll
                for (int ti = 0; ti < 2; ++ti)
#pragma unroll
                    for (int tj = 0; tj < 4; ++tj)
                        acc[ti][tj] = __builtin_amdgcn_wmma_f32_16x16x32_bf16(
                            false, fa[ti].v, false, fb[tj].v,
                            (short)0, acc[ti][tj], false, false);
            }
        }

        // per-row argmax over this j-tile (C layout: lane N = col, row = 8*half + r)
        const int jbase = jt + waveCol * 64 + N;
#pragma unroll
        for (int ti = 0; ti < 2; ++ti) {
#pragma unroll
            for (int r = 0; r < 8; ++r) {
                float v = acc[ti][0][r];
                int   j = jbase;
#pragma unroll
                for (int tj = 1; tj < 4; ++tj) {
                    float vc = acc[ti][tj][r];
                    int   jc = jbase + tj * 16;
                    if (vc > v || (vc == v && jc < j)) { v = vc; j = jc; }
                }
#pragma unroll
                for (int m = 1; m < 16; m <<= 1) {  // stay within 16-lane half
                    float ov = __shfl_xor(v, m, 32);
                    int   oj = __shfl_xor(j, m, 32);
                    if (ov > v || (ov == v && oj < j)) { v = ov; j = oj; }
                }
                int t = ti * 8 + r;
                if (v > bestv[t] || (v == bestv[t] && j < besti[t])) {
                    bestv[t] = v; besti[t] = j;
                }
            }
        }
    }

    // merge across the 4 wave-columns (same i rows, disjoint j ranges)
    if (N == 0) {
#pragma unroll
        for (int ti = 0; ti < 2; ++ti)
#pragma unroll
            for (int r = 0; r < 8; ++r) {
                int row = waveRow * 32 + ti * 16 + half * 8 + r;
                mV[waveCol][row] = bestv[ti * 8 + r];
                mI[waveCol][row] = besti[ti * 8 + r];
            }
    }
    __syncthreads();
    if (tid < BI) {
        float v = mV[0][tid];
        int   j = mI[0][tid];
#pragma unroll
        for (int w = 1; w < 4; ++w) {
            float ov = mV[w][tid];
            int   oj = mI[w][tid];
            if (ov > v || (ov == v && oj < j)) { v = ov; j = oj; }
        }
        zbest[i0 + tid] = j;
    }
}

// ---------------- kernel 4: zero the scalar output ----------------
__global__ void init_out_kernel(float* out) { out[0] = 0.f; }

// -------- kernel 5: exact f32 cosine loss with gathered columns --------
__global__ __launch_bounds__(256) void loss_kernel(
        const float* __restrict__ a, const float* __restrict__ b,
        const int* __restrict__ z, float* __restrict__ out) {
    __shared__ float sd[256], sa[256], sb[256];
    const int i = blockIdx.x;
    const int zi = z[i];
    float dot = 0.f, na = 0.f, nb = 0.f;
    for (int c = threadIdx.x; c < CCH; c += 256) {
        float av = a[(size_t)c * HW + i];
        float bv = b[(size_t)c * HW + zi];
        dot += av * bv; na += av * av; nb += bv * bv;
    }
    sd[threadIdx.x] = dot; sa[threadIdx.x] = na; sb[threadIdx.x] = nb;
    __syncthreads();
    for (int st = 128; st > 0; st >>= 1) {
        if (threadIdx.x < st) {
            sd[threadIdx.x] += sd[threadIdx.x + st];
            sa[threadIdx.x] += sa[threadIdx.x + st];
            sb[threadIdx.x] += sb[threadIdx.x + st];
        }
        __syncthreads();
    }
    if (threadIdx.x == 0) {
        float cs = sd[0] / ((sqrtf(sa[0]) + EPSF) * (sqrtf(sb[0]) + EPSF));
        atomicAdd(out, (1.f - cs) * (1.f / (float)HW));
    }
}

extern "C" void kernel_launch(void* const* d_in, const int* in_sizes, int n_in,
                              void* d_out, int out_size, void* d_ws, size_t ws_size,
                              hipStream_t stream) {
    const float* x = (const float*)d_in[0];   // [768][16384] f32
    const float* s = (const float*)d_in[1];   // [768][16384] f32
    float* out = (float*)d_out;
    char* ws = (char*)d_ws;

    // workspace layout (bytes)
    const size_t SZ_BNORM = (size_t)HW * 4;                 //   64 KB
    const size_t SZ_BF    = (size_t)HW * CCH * 2;           // 24 MB each
    float*          bnorm = (float*)ws;
    unsigned short* At    = (unsigned short*)(ws + SZ_BNORM);
    unsigned short* Bt    = (unsigned short*)(ws + SZ_BNORM + SZ_BF);
    int*            zbest = (int*)(ws + SZ_BNORM + 2 * SZ_BF);

    bnorm_kernel<<<HW / 256, 256, 0, stream>>>(s, bnorm);
    transpose_bf16_kernel<<<dim3(HW / 32, CCH / 32), 256, 0, stream>>>(x, bnorm, 0, At);
    transpose_bf16_kernel<<<dim3(HW / 32, CCH / 32), 256, 0, stream>>>(s, bnorm, 1, Bt);
    gemm_argmax_kernel<<<HW / BI, 256, 0, stream>>>(At, Bt, zbest);
    init_out_kernel<<<1, 1, 0, stream>>>(out);
    loss_kernel<<<HW, 256, 0, stream>>>(x, s, zbest, out);
}
